// CVIM_9363028706046
// MI455X (gfx1250) — matrix operand order repaired and checked by reading, not compiled, not run on hardware
//
#include <hip/hip_runtime.h>
#include <hip/hip_bf16.h>

typedef __attribute__((ext_vector_type(16))) _Float16 v16h;
typedef __attribute__((ext_vector_type(8)))  float    v8f;

#define C_    64
#define B_    2
#define H_    128
#define W_    512
#define HW_   65536      // H*W
#define SCALE_ 0.125f    // 64^-0.5

union Acc { v8f v; float e[8]; };
union BF  { v16h v; _Float16 e[16]; };

// ---- CDNA5 async global->LDS copy helpers -------------------------------
__device__ __forceinline__ uint32_t lds_addr(const void* p) {
  return (uint32_t)(uintptr_t)p;   // low 32 bits of LDS-aperture flat ptr
}
__device__ __forceinline__ void async_b128(uint32_t lds_off, const void* g) {
  asm volatile("global_load_async_to_lds_b128 %0, %1, off"
               :: "v"(lds_off), "v"(g) : "memory");
}
__device__ __forceinline__ void wait_async0() {
  asm volatile("s_wait_asynccnt 0" ::: "memory");
}

// ---------------------------------------------------------------------------
// one-shot: pack a 64x64 f32 weight matrix into f16 WMMA A-fragment layout
// ---------------------------------------------------------------------------
__global__ void k_pack_w(const float* __restrict__ w, _Float16* __restrict__ wp) {
  const int idx  = blockIdx.x * 256 + threadIdx.x;   // < 4096
  const int j    = idx & 15;
  const int lane = (idx >> 4) & 31;
  const int fi   = idx >> 9;                          // mt*2 + ks
  const int ks   = fi & 1, mt = fi >> 1;
  const int hi16 = (lane >= 16) ? 1 : 0;
  const int o = mt * 16 + (lane & 15);
  const int k = ks * 32 + j + (hi16 ? 16 : 0);
  wp[idx] = (_Float16)w[o * C_ + k];
}

// ---------------------------------------------------------------------------
// one-shot: pack x (f32 [B,C,H,W]) into conv B-fragment f16 layout
// xb[((pt*2+ks)*32 + lane)*16 + j], pt = 16-pixel tile, k = channel
// ---------------------------------------------------------------------------
__global__ void k_pack_x(const float* __restrict__ x, _Float16* __restrict__ xb) {
  const int idx = blockIdx.x * 256 + threadIdx.x;     // B*C*H*W
  const int p = idx & (HW_ - 1);
  const int i = (idx >> 16) & (C_ - 1);
  const int b = idx >> 22;
  const int pt = (b * HW_ + p) >> 4;
  const int ks = i >> 5, i5 = i & 31;
  const int hi = i5 >> 4, j = i5 & 15;
  const int lane = (p & 15) + 16 * hi;
  xb[(size_t)((pt * 2 + ks) * 32 + lane) * 16 + j] = (_Float16)x[idx];
}

// ---------------------------------------------------------------------------
// 1x1 conv: pure WMMA, all operands pre-packed (no LDS needed)
// ---------------------------------------------------------------------------
__global__ void k_conv1x1(const _Float16* __restrict__ xb, const _Float16* __restrict__ wp,
                          const float* __restrict__ bias, float* __restrict__ y) {
  const int lane = threadIdx.x & 31;
  const int wv   = threadIdx.x >> 5;
  const int hi16 = (lane >= 16) ? 1 : 0;
  const int pt   = blockIdx.x * 8 + wv;
  const int npix = pt * 16 + (lane & 15);
  const int b = npix >> 16, p = npix & (HW_ - 1);
  const v16h* xfr = (const v16h*)xb;
  const v16h* wfr = (const v16h*)wp;

  v16h bf0 = xfr[(pt * 2 + 0) * 32 + lane];
  v16h bf1 = xfr[(pt * 2 + 1) * 32 + lane];

#pragma unroll
  for (int mt = 0; mt < 4; ++mt) {
    Acc acc;
#pragma unroll
    for (int r = 0; r < 8; ++r)
      acc.e[r] = bias[mt * 16 + r + (hi16 ? 8 : 0)];
    v16h a0 = wfr[(mt * 2 + 0) * 32 + lane];
    v16h a1 = wfr[(mt * 2 + 1) * 32 + lane];
    acc.v = __builtin_amdgcn_wmma_f32_16x16x32_f16(false, a0, false, bf0,
                                                   (short)0, acc.v, false, false);
    acc.v = __builtin_amdgcn_wmma_f32_16x16x32_f16(false, a1, false, bf1,
                                                   (short)0, acc.v, false, false);
#pragma unroll
    for (int r = 0; r < 8; ++r) {
      int o = mt * 16 + r + (hi16 ? 8 : 0);
      y[(size_t)(b * C_ + o) * HW_ + p] = acc.e[r];
    }
  }
}

// ---------------------------------------------------------------------------
// depthwise 3x3 + bias (+scale); scatters f16 output into consumer fragment
// layout. mode 0: logits-A (m=w,k=c); 1: logits-B (k=c,n=v); 2: agg-A (m=c,k=v)
// ---------------------------------------------------------------------------
__global__ void k_dwconv(const float* __restrict__ t, const float* __restrict__ wd,
                         const float* __restrict__ bd, _Float16* __restrict__ q,
                         float oscale, int mode) {
  const int idx = blockIdx.x * 256 + threadIdx.x;   // B*C*H*W
  const int w = idx & (W_ - 1);
  const int h = (idx >> 9) & (H_ - 1);
  const int c = (idx >> 16) & (C_ - 1);
  const int b = idx >> 22;
  const float* tc = t + (size_t)(b * C_ + c) * HW_;
  float s = bd[c];
#pragma unroll
  for (int dy = 0; dy < 3; ++dy) {
    int hh = h + dy - 1;
    if (hh < 0 || hh >= H_) continue;
#pragma unroll
    for (int dx = 0; dx < 3; ++dx) {
      int ww = w + dx - 1;
      if (ww < 0 || ww >= W_) continue;
      s += tc[hh * W_ + ww] * wd[c * 9 + dy * 3 + dx];
    }
  }
  const int bh = b * H_ + h;
  size_t off;
  if (mode == 0) {            // A-frag over (m=w, k=c)
    int wt = w >> 4, ks = c >> 5, c5 = c & 31;
    int hi = (c5 >> 4) & 1, j = c5 & 15;
    int lane = (w & 15) + 16 * hi;
    off = (size_t)(((bh * 32 + wt) * 2 + ks) * 32 + lane) * 16 + j;
  } else if (mode == 1) {     // B-frag over (k=c, n=v)
    int vt = w >> 4, ks = c >> 5;
    int hi = (c >> 4) & 1, j = c & 15;
    int lane = (w & 15) + 16 * hi;
    off = (size_t)(((bh * 32 + vt) * 2 + ks) * 32 + lane) * 16 + j;
  } else {                    // A-frag over (m=c, k=v/w)
    int mt = c >> 4, ks = w >> 5, k5 = w & 31;
    int hi = (k5 >> 4) & 1, j = k5 & 15;
    int lane = (c & 15) + 16 * hi;
    off = (size_t)(((bh * 4 + mt) * 16 + ks) * 32 + lane) * 16 + j;
  }
  q[off] = (_Float16)(s * oscale);
}

// ---------------------------------------------------------------------------
// logits: att[bh,w,v] = sum_c Qls*Qr  (SCALE folded into Ql)
// packed 16KB slices staged async->LDS; all fragments contiguous v16h
// ---------------------------------------------------------------------------
__global__ void k_logits(const _Float16* __restrict__ QlA, const _Float16* __restrict__ QrB,
                         _Float16* __restrict__ att) {
  __shared__ _Float16 sL[8192];                    // 16 KB
  __shared__ _Float16 sR[8192];                    // 16 KB
  const int tid  = threadIdx.x;
  const int lane = tid & 31;
  const int wv   = tid >> 5;
  const int hi16 = (lane >= 16) ? 1 : 0;
  const int bh   = blockIdx.z;
  const int w0   = blockIdx.x * 128;
  const int v0   = blockIdx.y * 128;
  const _Float16* la = QlA + (size_t)bh * 32768 + (size_t)blockIdx.x * 8192;
  const _Float16* rb = QrB + (size_t)bh * 32768 + (size_t)blockIdx.y * 8192;

#pragma unroll
  for (int it = 0; it < 4; ++it) {
    int t = tid + it * 256;                        // 1024 chunks per buffer
    async_b128(lds_addr(&sL[t * 8]), la + t * 8);
    async_b128(lds_addr(&sR[t * 8]), rb + t * 8);
  }
  wait_async0();
  __syncthreads();

  const v16h* sLv = (const v16h*)sL;
  const v16h* sRv = (const v16h*)sR;
  v16h a0 = sLv[(wv * 2 + 0) * 32 + lane];
  v16h a1 = sLv[(wv * 2 + 1) * 32 + lane];

#pragma unroll
  for (int nt = 0; nt < 8; ++nt) {
    Acc acc;
#pragma unroll
    for (int r = 0; r < 8; ++r) acc.e[r] = 0.0f;
    v16h b0 = sRv[(nt * 2 + 0) * 32 + lane];
    v16h b1 = sRv[(nt * 2 + 1) * 32 + lane];
    acc.v = __builtin_amdgcn_wmma_f32_16x16x32_f16(false, a0, false, b0,
                                                   (short)0, acc.v, false, false);
    acc.v = __builtin_amdgcn_wmma_f32_16x16x32_f16(false, a1, false, b1,
                                                   (short)0, acc.v, false, false);
#pragma unroll
    for (int r = 0; r < 8; ++r) {
      int wr = w0 + wv * 16 + r + (hi16 ? 8 : 0);
      int vc = v0 + nt * 16 + (lane & 15);
      att[(size_t)bh * W_ * W_ + (size_t)wr * W_ + vc] = (_Float16)acc.e[r];
    }
  }
}

// ---------------------------------------------------------------------------
// softmax stats
// ---------------------------------------------------------------------------
__global__ void k_rowstats(const _Float16* __restrict__ att,
                           float* __restrict__ rmax, float* __restrict__ rsum) {
  const int idx = blockIdx.x * 256 + threadIdx.x;   // bh*512 + w
  const _Float16* row = att + (size_t)idx * W_;
  float m = -1e30f;
  for (int v = 0; v < W_; ++v) m = fmaxf(m, (float)row[v]);
  float s = 0.0f;
  for (int v = 0; v < W_; ++v) s += __expf((float)row[v] - m);
  rmax[idx] = m;
  rsum[idx] = s;
}

__global__ void k_colstats(const _Float16* __restrict__ att,
                           float* __restrict__ cmax, float* __restrict__ csum) {
  const int idx = blockIdx.x * 256 + threadIdx.x;   // bh*512 + v
  const int bh = idx >> 9;
  const int v  = idx & (W_ - 1);
  const _Float16* a = att + (size_t)bh * W_ * W_ + v;
  float m = -1e30f;
  for (int w = 0; w < W_; ++w) m = fmaxf(m, (float)a[(size_t)w * W_]);
  float s = 0.0f;
  for (int w = 0; w < W_; ++w) s += __expf((float)a[(size_t)w * W_] - m);
  cmax[idx] = m;
  csum[idx] = s;
}

// ---------------------------------------------------------------------------
// F_r2l[c,w] = (1/rsum[w]) * sum_v Vr[c,v] * exp(att[w,v]-rmax[w])
// A = packed Vr (64KB LDS, contiguous v16h); output packed for k_final
// ---------------------------------------------------------------------------
__global__ void k_agg_r2l(const _Float16* __restrict__ att, const _Float16* __restrict__ VrA,
                          const float* __restrict__ rmax, const float* __restrict__ rsum,
                          _Float16* __restrict__ XF1) {
  __shared__ _Float16 sV[32768];                   // 64 KB
  const int tid  = threadIdx.x;
  const int lane = tid & 31;
  const int wv   = tid >> 5;
  const int hi16 = (lane >= 16) ? 1 : 0;
  const int bh   = blockIdx.y;
  const _Float16* va = VrA + (size_t)bh * 32768;

#pragma unroll
  for (int it = 0; it < 16; ++it) {                // 4096 chunks of 16B
    int t = tid + it * 256;
    async_b128(lds_addr(&sV[t * 8]), va + t * 8);
  }
  wait_async0();
  __syncthreads();

  const v16h* sVv = (const v16h*)sV;
  const int w0 = (blockIdx.x * 8 + wv) * 16;
  const int wl = w0 + (lane & 15);
  const _Float16* arow = att + (size_t)bh * W_ * W_ + (size_t)wl * W_;
  const float rm  = rmax[bh * W_ + wl];
  const float inv = 1.0f / rsum[bh * W_ + wl];

  Acc acc[4];
#pragma unroll
  for (int mt = 0; mt < 4; ++mt)
#pragma unroll
    for (int r = 0; r < 8; ++r) acc[mt].e[r] = 0.0f;

  for (int ks = 0; ks < 16; ++ks) {
    BF bf;                                         // B[k=v, n=w] = P[w,v]
#pragma unroll
    for (int j = 0; j < 16; ++j) {
      int k = ks * 32 + j + (hi16 ? 16 : 0);
      bf.e[j] = (_Float16)__expf((float)arow[k] - rm);
    }
#pragma unroll
    for (int mt = 0; mt < 4; ++mt) {
      v16h a = sVv[(mt * 16 + ks) * 32 + lane];
      acc[mt].v = __builtin_amdgcn_wmma_f32_16x16x32_f16(false, a, false, bf.v,
                                                         (short)0, acc[mt].v, false, false);
    }
  }
  const int pt = bh * 32 + (w0 >> 4);              // pixel tile
#pragma unroll
  for (int mt = 0; mt < 4; ++mt) {
    int ksq = mt >> 1;
    int laneq = (lane & 15) + 16 * (mt & 1);
#pragma unroll
    for (int r = 0; r < 8; ++r) {
      int jq = r + 8 * hi16;
      XF1[(size_t)(((pt * 2 + ksq)) * 32 + laneq) * 16 + jq] =
          (_Float16)(acc[mt].e[r] * inv);
    }
  }
}

// ---------------------------------------------------------------------------
// F_l2r[c,v] = (1/csum[v]) * sum_w Vl[c,w] * exp(att[w,v]-cmax[v])
// ---------------------------------------------------------------------------
__global__ void k_agg_l2r(const _Float16* __restrict__ att, const _Float16* __restrict__ VlA,
                          const float* __restrict__ cmax, const float* __restrict__ csum,
                          _Float16* __restrict__ XF2) {
  __shared__ _Float16 sV[32768];                   // 64 KB
  const int tid  = threadIdx.x;
  const int lane = tid & 31;
  const int wv   = tid >> 5;
  const int hi16 = (lane >= 16) ? 1 : 0;
  const int bh   = blockIdx.y;
  const _Float16* va = VlA + (size_t)bh * 32768;

#pragma unroll
  for (int it = 0; it < 16; ++it) {
    int t = tid + it * 256;
    async_b128(lds_addr(&sV[t * 8]), va + t * 8);
  }
  wait_async0();
  __syncthreads();

  const v16h* sVv = (const v16h*)sV;
  const int v0  = (blockIdx.x * 8 + wv) * 16;
  const int vlc = v0 + (lane & 15);
  const _Float16* acol = att + (size_t)bh * W_ * W_ + vlc;
  const float cm  = cmax[bh * W_ + vlc];
  const float inv = 1.0f / csum[bh * W_ + vlc];

  Acc acc[4];
#pragma unroll
  for (int mt = 0; mt < 4; ++mt)
#pragma unroll
    for (int r = 0; r < 8; ++r) acc[mt].e[r] = 0.0f;

  for (int ks = 0; ks < 16; ++ks) {
    BF bf;                                         // B[k=w, n=v] = P2[w,v]
#pragma unroll
    for (int j = 0; j < 16; ++j) {
      int k = ks * 32 + j + (hi16 ? 16 : 0);
      bf.e[j] = (_Float16)__expf((float)acol[(size_t)k * W_] - cm);
    }
#pragma unroll
    for (int mt = 0; mt < 4; ++mt) {
      v16h a = sVv[(mt * 16 + ks) * 32 + lane];
      acc[mt].v = __builtin_amdgcn_wmma_f32_16x16x32_f16(false, a, false, bf.v,
                                                         (short)0, acc[mt].v, false, false);
    }
  }
  const int pt = bh * 32 + (v0 >> 4);
#pragma unroll
  for (int mt = 0; mt < 4; ++mt) {
    int ksq = mt >> 1;
    int laneq = (lane & 15) + 16 * (mt & 1);
#pragma unroll
    for (int r = 0; r < 8; ++r) {
      int jq = r + 8 * hi16;
      XF2[(size_t)(((pt * 2 + ksq)) * 32 + laneq) * 16 + jq] =
          (_Float16)(acc[mt].e[r] * inv);
    }
  }
}

// ---------------------------------------------------------------------------
// final: out = x_l + x_r + (lp3_w @ F1 + lp3_b) + (rp3_w @ F2 + rp3_b)
// all operands pre-packed; no LDS
// ---------------------------------------------------------------------------
__global__ void k_final(const _Float16* __restrict__ XF1, const _Float16* __restrict__ XF2,
                        const _Float16* __restrict__ wp1, const float* __restrict__ b1,
                        const _Float16* __restrict__ wp2, const float* __restrict__ b2,
                        const float* __restrict__ xl, const float* __restrict__ xr,
                        float* __restrict__ out) {
  const int lane = threadIdx.x & 31;
  const int wv   = threadIdx.x >> 5;
  const int hi16 = (lane >= 16) ? 1 : 0;
  const int pt   = blockIdx.x * 8 + wv;
  const int npix = pt * 16 + (lane & 15);
  const int b = npix >> 16, p = npix & (HW_ - 1);
  const v16h* f1 = (const v16h*)XF1;
  const v16h* f2 = (const v16h*)XF2;
  const v16h* w1fr = (const v16h*)wp1;
  const v16h* w2fr = (const v16h*)wp2;

  v16h f10 = f1[(pt * 2 + 0) * 32 + lane];
  v16h f11 = f1[(pt * 2 + 1) * 32 + lane];
  v16h f20 = f2[(pt * 2 + 0) * 32 + lane];
  v16h f21 = f2[(pt * 2 + 1) * 32 + lane];

#pragma unroll
  for (int mt = 0; mt < 4; ++mt) {
    Acc acc;
#pragma unroll
    for (int r = 0; r < 8; ++r) {
      int o = mt * 16 + r + (hi16 ? 8 : 0);
      acc.e[r] = b1[o] + b2[o];
    }
    v16h a10 = w1fr[(mt * 2 + 0) * 32 + lane];
    v16h a11 = w1fr[(mt * 2 + 1) * 32 + lane];
    v16h a20 = w2fr[(mt * 2 + 0) * 32 + lane];
    v16h a21 = w2fr[(mt * 2 + 1) * 32 + lane];
    acc.v = __builtin_amdgcn_wmma_f32_16x16x32_f16(false, a10, false, f10,
                                                   (short)0, acc.v, false, false);
    acc.v = __builtin_amdgcn_wmma_f32_16x16x32_f16(false, a11, false, f11,
                                                   (short)0, acc.v, false, false);
    acc.v = __builtin_amdgcn_wmma_f32_16x16x32_f16(false, a20, false, f20,
                                                   (short)0, acc.v, false, false);
    acc.v = __builtin_amdgcn_wmma_f32_16x16x32_f16(false, a21, false, f21,
                                                   (short)0, acc.v, false, false);
#pragma unroll
    for (int r = 0; r < 8; ++r) {
      int o = mt * 16 + r + (hi16 ? 8 : 0);
      size_t idx = (size_t)(b * C_ + o) * HW_ + p;
      out[idx] = acc.e[r] + xl[idx] + xr[idx];
    }
  }
}

// ---------------------------------------------------------------------------
extern "C" void kernel_launch(void* const* d_in, const int* in_sizes, int n_in,
                              void* d_out, int out_size, void* d_ws, size_t ws_size,
                              hipStream_t stream) {
  (void)in_sizes; (void)n_in; (void)out_size; (void)ws_size;
  const float* x_l = (const float*)d_in[0];
  const float* x_r = (const float*)d_in[1];
  const float* lp1_w1 = (const float*)d_in[2];
  const float* lp1_b1 = (const float*)d_in[3];
  const float* lp1_wd = (const float*)d_in[4];
  const float* lp1_bd = (const float*)d_in[5];
  const float* rp1_w1 = (const float*)d_in[6];
  const float* rp1_b1 = (const float*)d_in[7];
  const float* rp1_wd = (const float*)d_in[8];
  const float* rp1_bd = (const float*)d_in[9];
  const float* lp2_w1 = (const float*)d_in[10];
  const float* lp2_b1 = (const float*)d_in[11];
  const float* lp2_wd = (const float*)d_in[12];
  const float* lp2_bd = (const float*)d_in[13];
  const float* rp2_w1 = (const float*)d_in[14];
  const float* rp2_b1 = (const float*)d_in[15];
  const float* rp2_wd = (const float*)d_in[16];
  const float* rp2_bd = (const float*)d_in[17];
  const float* lp3_w  = (const float*)d_in[18];
  const float* lp3_b  = (const float*)d_in[19];
  const float* rp3_w  = (const float*)d_in[20];
  const float* rp3_b  = (const float*)d_in[21];

  char* ws = (char*)d_ws;
  const size_t MB = 1024ull * 1024ull;
  _Float16* QlA  = (_Float16*)(ws + 0 * MB);    // 16 MB each (packed)
  _Float16* QrB  = (_Float16*)(ws + 16 * MB);
  _Float16* VlA  = (_Float16*)(ws + 32 * MB);
  _Float16* VrA  = (_Float16*)(ws + 48 * MB);
  _Float16* att  = (_Float16*)(ws + 64 * MB);   // 128 MB
  float*    tconv= (float*)(ws + 192 * MB);     // 32 MB, reused per path
  _Float16* XF1  = (_Float16*)(ws + 224 * MB);  // 16 MB (packed)
  _Float16* XF2  = (_Float16*)(ws + 240 * MB);  // 16 MB (packed)
  _Float16* XBl  = (_Float16*)(ws + 256 * MB);  // 16 MB (packed x_l)
  _Float16* XBr  = (_Float16*)(ws + 272 * MB);  // 16 MB (packed x_r)
  float* rmaxb   = (float*)(ws + 288 * MB);     // 4 x 512 KB stats
  float* rsumb   = rmaxb + 131072;
  float* cmaxb   = rsumb + 131072;
  float* csumb   = cmaxb + 131072;
  _Float16* wpk  = (_Float16*)(ws + 291 * MB);  // 6 x 8 KB packed weights
  _Float16* wp_l1 = wpk + 0 * 4096;
  _Float16* wp_r1 = wpk + 1 * 4096;
  _Float16* wp_l2 = wpk + 2 * 4096;
  _Float16* wp_r2 = wpk + 3 * 4096;
  _Float16* wp_l3 = wpk + 4 * 4096;
  _Float16* wp_r3 = wpk + 5 * 4096;

  const dim3 blk(256);

  // one-shot preprocessing
  k_pack_w<<<16, blk, 0, stream>>>(lp1_w1, wp_l1);
  k_pack_w<<<16, blk, 0, stream>>>(rp1_w1, wp_r1);
  k_pack_w<<<16, blk, 0, stream>>>(lp2_w1, wp_l2);
  k_pack_w<<<16, blk, 0, stream>>>(rp2_w1, wp_r2);
  k_pack_w<<<16, blk, 0, stream>>>(lp3_w,  wp_l3);
  k_pack_w<<<16, blk, 0, stream>>>(rp3_w,  wp_r3);
  k_pack_x<<<32768, blk, 0, stream>>>(x_l, XBl);
  k_pack_x<<<32768, blk, 0, stream>>>(x_r, XBr);

  // projections (Ql gets SCALE folded in); dwconv scatters into packed layouts
  struct Path { const _Float16 *xb, *wp; const float *b1, *wd, *bd;
                _Float16* q; float sc; int mode; };
  const Path paths[4] = {
      {XBl, wp_l1, lp1_b1, lp1_wd, lp1_bd, QlA, SCALE_, 0},
      {XBr, wp_r1, rp1_b1, rp1_wd, rp1_bd, QrB, 1.0f,   1},
      {XBl, wp_l2, lp2_b1, lp2_wd, lp2_bd, VlA, 1.0f,   2},
      {XBr, wp_r2, rp2_b1, rp2_wd, rp2_bd, VrA, 1.0f,   2}};
  for (int i = 0; i < 4; ++i) {
    k_conv1x1<<<1024, blk, 0, stream>>>(paths[i].xb, paths[i].wp, paths[i].b1, tconv);
    k_dwconv<<<32768, blk, 0, stream>>>(tconv, paths[i].wd, paths[i].bd,
                                        paths[i].q, paths[i].sc, paths[i].mode);
  }

  k_logits<<<dim3(4, 4, 256), blk, 0, stream>>>(QlA, QrB, att);
  k_rowstats<<<512, blk, 0, stream>>>(att, rmaxb, rsumb);
  k_colstats<<<512, blk, 0, stream>>>(att, cmaxb, csumb);
  k_agg_r2l<<<dim3(4, 256), blk, 0, stream>>>(att, VrA, rmaxb, rsumb, XF1);
  k_agg_l2r<<<dim3(4, 256), blk, 0, stream>>>(att, VlA, cmaxb, csumb, XF2);
  k_final<<<1024, blk, 0, stream>>>(XF1, XF2, wp_l3, lp3_b, wp_r3, rp3_b,
                                    x_l, x_r, (float*)d_out);
}
